// FullyFusedMLP_34857954574419
// MI455X (gfx1250) — matrix operand order repaired and checked by reading, not compiled, not run on hardware
//
#include <hip/hip_runtime.h>

typedef __attribute__((ext_vector_type(16))) _Float16 v16h;
typedef __attribute__((ext_vector_type(8)))  _Float16 h8;
typedef __attribute__((ext_vector_type(8)))  float    v8f;
typedef __attribute__((ext_vector_type(8)))  float    f8;

#define IN_DIMS         32
#define HIDDEN          64
#define N_INNER         4
#define OUT_DIMS        3
#define PTS_PER_WAVE    32          // two 16-point tiles per wave
#define WAVES_PER_BLOCK 8
#define PTS_PER_BLOCK   (PTS_PER_WAVE * WAVES_PER_BLOCK)    // 256

// ws image layout (in halves)
#define WHID_IMG   0                // 4*64*64 = 16384 halves (swizzled, row-permuted)
#define WIN_IMG    16384            // 4 ctiles * 32 lanes * 16 halves = 2048
#define WOUT_IMG   (16384 + 2048)   // 2 kchunks * 32 lanes * 16 halves = 1024
#define BIAS_HOFF  (16384 + 2048 + 1024)   // then 328 f32: b_in|b_hid|b_out(padded 8)
#define N_BIAS_F32     328
#define N_WHID_CHUNKS  (N_INNER * HIDDEN * 8)   // 2048 x 16B
#define N_BIAS_CHUNKS  (N_BIAS_F32 / 4)          // 82 x 16B
#define N_DMA_CHUNKS   (N_WHID_CHUNKS + N_BIAS_CHUNKS)

// ---- XOR bank swizzle on 16B chunks ----
__device__ __forceinline__ int swz(int row) { return (row ^ (row >> 3)) & 7; }
__device__ __forceinline__ int chunk_idx(int row, int lc) {   // half-index
    return row * HIDDEN + ((lc ^ swz(row)) << 3);
}
// A-matrix fragment K index (contiguous, matches B): k = hi*16 + e
__device__ __forceinline__ int frag_col(int e, int hi) {
    return hi * 16 + e;
}
// Weight-row permutation: tile c, tile-row t holds weight row pi(c,t), so that
// D (m = r + 8*hi) packs within-lane into the next layer's B (k = hi*16 + e).
__device__ __forceinline__ int pi_row(int c, int t) {
    return (c & 1) * 8 + (c >> 1) * 32 + (t & 7) + 16 * (t >> 3);
}
// bias f8 offset for D-tile c on lane-half hi (dims pi(c, r+8hi), r=0..7 contiguous)
__device__ __forceinline__ int bias_off(int c, int hi) {
    return (c & 1) * 8 + (c >> 1) * 32 + 16 * hi;
}
// f32->f16 pack of one 8-row accumulator slice (bias already inside the acc)
__device__ __forceinline__ h8 cvt_h8(const v8f a) {
    h8 o;
    #pragma unroll
    for (int r = 0; r < 8; ++r) o[r] = (_Float16)a[r];
    return o;
}

// =====================================================================
// Pre-pass (once per launch): fp32 weights -> f16 fragment/LDS images + bias image
// =====================================================================
__global__ __launch_bounds__(256)
void FFMLP_prep_weights(const float* __restrict__ W_in,
                        const float* __restrict__ b_in,
                        const float* __restrict__ W_hid,
                        const float* __restrict__ b_hid,
                        const float* __restrict__ W_out,
                        const float* __restrict__ b_out,
                        _Float16* __restrict__ ws)
{
    const int tid = blockIdx.x * 256 + threadIdx.x;
    const int stride = gridDim.x * 256;

    // (a) W_hid -> swizzled, row-permuted f16 image (byte-exact LDS layout)
    for (int ci = tid; ci < N_WHID_CHUNKS; ci += stride) {
        const int l  = ci >> 9;
        const int ir = (ci >> 3) & 63;          // image row = c*16 + t
        const int lc = ci & 7;
        const int src = pi_row(ir >> 4, ir & 15);
        const f8 w = *(const f8*)(W_hid + (l * HIDDEN + src) * HIDDEN + lc * 8);
        h8 hw;
        #pragma unroll
        for (int e = 0; e < 8; ++e) hw[e] = (_Float16)w[e];
        *(h8*)(ws + WHID_IMG + l * HIDDEN * HIDDEN + chunk_idx(ir, lc)) = hw;
    }
    // (b) W_in -> fragment-ready A image [c][lane][16], row-permuted
    for (int f = tid; f < 4 * 32 * 16; f += stride) {
        const int c = f >> 9, lane = (f >> 4) & 31, e = f & 15;
        const int l15 = lane & 15, hi = lane >> 4;
        ws[WIN_IMG + f] = (_Float16)W_in[pi_row(c, l15) * IN_DIMS + frag_col(e, hi)];
    }
    // (c) W_out (zero-padded 3->16 rows, identity order) -> A image [kc][lane][16]
    for (int f = tid; f < 2 * 32 * 16; f += stride) {
        const int kc = f >> 9, lane = (f >> 4) & 31, e = f & 15;
        const int o = lane & 15, hi = lane >> 4;
        ws[WOUT_IMG + f] = (o < OUT_DIMS)
            ? (_Float16)W_out[o * HIDDEN + kc * 32 + frag_col(e, hi)]
            : (_Float16)0.0f;
    }
    // (d) bias image f32: [b_in(64) | b_hid(256) | b_out padded to 8]
    float* bimg = (float*)(ws + BIAS_HOFF);
    for (int i = tid; i < N_BIAS_F32; i += stride) {
        float v;
        if (i < HIDDEN)                      v = b_in[i];
        else if (i < (1 + N_INNER) * HIDDEN) v = b_hid[i - HIDDEN];
        else                                 v = (i - (1 + N_INNER) * HIDDEN < OUT_DIMS)
                                                 ? b_out[i - (1 + N_INNER) * HIDDEN] : 0.0f;
        bimg[i] = v;
    }
}

// =====================================================================
// Main fully-fused MLP: Y = W x H, activations live in registers as B
// fragments (points in lanes); biases preloaded into WMMA C operands.
// =====================================================================
__global__ __launch_bounds__(256)
void FullyFusedMLP_34857954574419_kernel(
        const float* __restrict__ x,
        const _Float16* __restrict__ ws,
        float* __restrict__ out)
{
    __shared__ __align__(32) _Float16 sWhid[N_INNER * HIDDEN * HIDDEN]; // 32KB, swizzled
    __shared__ __align__(32) float    sBias[N_BIAS_F32];                // 1.3KB

    const int tid  = threadIdx.x;
    const int wave = tid >> 5;           // wave32 on gfx1250
    const int lane = tid & 31;
    const int l15  = lane & 15;          // point index within a tile / A-row index
    const int hi   = lane >> 4;

    // ---- Async DMA weights + biases straight into LDS (ASYNCcnt-tracked) ----
    {
        const uint32_t whid_base = (uint32_t)(uintptr_t)(&sWhid[0]);
        const uint32_t bias_base = (uint32_t)(uintptr_t)(&sBias[0]);
        const char* gbase = (const char*)(ws);
        #pragma unroll
        for (int it = 0; it < (N_DMA_CHUNKS + 255) / 256; ++it) {
            const int ci = it * 256 + tid;
            if (ci < N_DMA_CHUNKS) {
                uint32_t lofs;
                const void* gaddr;
                if (ci < N_WHID_CHUNKS) {
                    lofs  = whid_base + ci * 16;
                    gaddr = (const void*)(gbase + ci * 16);
                } else {
                    const int j = ci - N_WHID_CHUNKS;
                    lofs  = bias_base + j * 16;
                    gaddr = (const void*)(gbase + BIAS_HOFF * 2 + j * 16);
                }
                asm volatile("global_load_async_to_lds_b128 %0, %1, off"
                             :: "v"(lofs), "v"(gaddr) : "memory");
            }
        }
    }

    const long long rowBase = (long long)blockIdx.x * PTS_PER_BLOCK
                            + (long long)wave * PTS_PER_WAVE;
    const v16h* wsWin  = (const v16h*)(ws + WIN_IMG);    // [c*32 + lane]
    const v16h* wsWout = (const v16h*)(ws + WOUT_IMG);   // [kc*32 + lane]
    const h8 hzero = {};

    // ---- Load layer-0 B fragments from x while DMA is in flight ----
    // B layout: element e <-> k = hi*16 + e  => 16 contiguous floats per lane.
    v16h B[2][2];                       // [pt][kc] activation fragments (registers!)
    v16h b0[2];
    #pragma unroll
    for (int pt = 0; pt < 2; ++pt) {
        const float* xp = x + (rowBase + pt * 16 + l15) * IN_DIMS + hi * 16;
        const f8 xa = *(const f8*)(xp);
        const f8 xb = *(const f8*)(xp + 8);
        #pragma unroll
        for (int e = 0; e < 8; ++e) {
            b0[pt][e]     = (_Float16)xa[e];
            b0[pt][8 + e] = (_Float16)xb[e];
        }
    }

    // Precompute per-lane swizzled fragment byte offsets (layer base is an immediate).
    int fofs[4][2][2];
    #pragma unroll
    for (int c = 0; c < 4; ++c)
        #pragma unroll
        for (int kc = 0; kc < 2; ++kc)
            #pragma unroll
            for (int sl = 0; sl < 2; ++sl)
                fofs[c][kc][sl] = chunk_idx(c * 16 + l15, kc * 4 + hi * 2 + sl) * 2;

    asm volatile("s_wait_asynccnt 0x0" ::: "memory");
    __syncthreads();

    // ================= Layer 0: relu(W_in @ x^T + b_in), K = 32 =================
    {
        v8f acc0[2][4];
        #pragma unroll
        for (int c = 0; c < 4; ++c) {
            const v8f bias = *(const v8f*)(sBias + bias_off(c, hi));  // C := bias
            const v16h A = wsWin[c * 32 + lane];
            #pragma unroll
            for (int pt = 0; pt < 2; ++pt)
                acc0[pt][c] = __builtin_amdgcn_wmma_f32_16x16x32_f16(
                    false, A, false, b0[pt], (short)0, bias, false, false);
        }
        #pragma unroll
        for (int pt = 0; pt < 2; ++pt)
            #pragma unroll
            for (int kc = 0; kc < 2; ++kc) {
                const h8 lo = __builtin_elementwise_max(cvt_h8(acc0[pt][2 * kc]),     hzero);
                const h8 hh = __builtin_elementwise_max(cvt_h8(acc0[pt][2 * kc + 1]), hzero);
                B[pt][kc] = __builtin_shufflevector(lo, hh,
                    0, 1, 2, 3, 4, 5, 6, 7, 8, 9, 10, 11, 12, 13, 14, 15);
            }
    }

    // ============ 4 hidden layers: relu(W @ h + b), K = 64; all in registers =======
    #pragma unroll
    for (int l = 0; l < N_INNER; ++l) {
        const float* bl_base = sBias + HIDDEN + l * HIDDEN;
        const char* wl = (const char*)sWhid + l * HIDDEN * HIDDEN * 2;

        v8f acc[2][4];
        #pragma unroll
        for (int c = 0; c < 4; ++c) {
            const v8f bias = *(const v8f*)(bl_base + bias_off(c, hi)); // C := bias
            #pragma unroll
            for (int pt = 0; pt < 2; ++pt) acc[pt][c] = bias;
        }

        #pragma unroll
        for (int kc = 0; kc < 2; ++kc)
            #pragma unroll
            for (int c = 0; c < 4; ++c) {
                const h8 alo = *(const h8*)(wl + fofs[c][kc][0]);
                const h8 ahi = *(const h8*)(wl + fofs[c][kc][1]);
                const v16h A = __builtin_shufflevector(alo, ahi,
                    0, 1, 2, 3, 4, 5, 6, 7, 8, 9, 10, 11, 12, 13, 14, 15);
                #pragma unroll
                for (int pt = 0; pt < 2; ++pt)     // each A fragment feeds 2 WMMAs
                    acc[pt][c] = __builtin_amdgcn_wmma_f32_16x16x32_f16(
                        false, A, false, B[pt][kc], (short)0, acc[pt][c], false, false);
            }

        #pragma unroll
        for (int pt = 0; pt < 2; ++pt)
            #pragma unroll
            for (int kc = 0; kc < 2; ++kc) {
                const h8 lo = __builtin_elementwise_max(cvt_h8(acc[pt][2 * kc]),     hzero);
                const h8 hh = __builtin_elementwise_max(cvt_h8(acc[pt][2 * kc + 1]), hzero);
                B[pt][kc] = __builtin_shufflevector(lo, hh,
                    0, 1, 2, 3, 4, 5, 6, 7, 8, 9, 10, 11, 12, 13, 14, 15);
            }
    }

    // ========== Output layer (pre-padded A image, bias in C) + sigmoid ==========
    {
        const v8f bias = *(const v8f*)(sBias + (1 + N_INNER) * HIDDEN); // b_out padded
        v8f acco[2] = {bias, bias};
        #pragma unroll
        for (int kc = 0; kc < 2; ++kc) {
            const v16h A = wsWout[kc * 32 + lane];        // rows 3..15 pre-zeroed
            #pragma unroll
            for (int pt = 0; pt < 2; ++pt)
                acco[pt] = __builtin_amdgcn_wmma_f32_16x16x32_f16(
                    false, A, false, B[pt][kc], (short)0, acco[pt], false, false);
        }
        if (hi == 0) {                 // m = r (+8*hi): only hi==0 holds dims 0..2
            #pragma unroll
            for (int pt = 0; pt < 2; ++pt) {
                float* op = out + (rowBase + pt * 16 + l15) * OUT_DIMS;
                #pragma unroll
                for (int r = 0; r < OUT_DIMS; ++r)
                    op[r] = 1.0f / (1.0f + __expf(-acco[pt][r]));
            }
        }
    }
}

extern "C" void kernel_launch(void* const* d_in, const int* in_sizes, int n_in,
                              void* d_out, int out_size, void* d_ws, size_t ws_size,
                              hipStream_t stream) {
    const float* x     = (const float*)d_in[0];
    const float* W_in  = (const float*)d_in[1];
    const float* b_in  = (const float*)d_in[2];
    const float* W_hid = (const float*)d_in[3];
    const float* b_hid = (const float*)d_in[4];
    const float* W_out = (const float*)d_in[5];
    const float* b_out = (const float*)d_in[6];
    float* out = (float*)d_out;
    _Float16* ws = (_Float16*)d_ws;   // needs 40,224 bytes of scratch

    hipLaunchKernelGGL(FFMLP_prep_weights, dim3(8), dim3(256), 0, stream,
                       W_in, b_in, W_hid, b_hid, W_out, b_out, ws);

    const int n = in_sizes[0] / IN_DIMS;          // 1<<20
    dim3 grid(n / PTS_PER_BLOCK);                 // 4096 blocks
    dim3 block(256);                              // 8 wave32 waves
    hipLaunchKernelGGL(FullyFusedMLP_34857954574419_kernel, grid, block, 0, stream,
                       x, ws, out);
}
